// IntraStockCorrelation_32031866093745
// MI455X (gfx1250) — matrix operand order repaired and checked by reading, not compiled, not run on hardware
//
#include <hip/hip_runtime.h>

// ---------------------------------------------------------------------------
// Types for CDNA5 WMMA / TDM
// ---------------------------------------------------------------------------
typedef __attribute__((ext_vector_type(16))) __bf16 v16bf;
typedef __attribute__((ext_vector_type(8)))  float  v8f;
typedef __attribute__((ext_vector_type(2)))  float  v2f;
typedef unsigned int u32x4 __attribute__((ext_vector_type(4)));
typedef int          i32x8 __attribute__((ext_vector_type(8)));
typedef int          i32x4 __attribute__((ext_vector_type(4)));

#define HDIM 512
#define NSTK 4096
#define KSUB 6

#if defined(__has_builtin)
#if __has_builtin(__builtin_amdgcn_tensor_load_to_lds) && __has_builtin(__builtin_amdgcn_s_wait_tensorcnt)
#define HAVE_TDM 1
#endif
#endif

__device__ __forceinline__ unsigned short f2bf(float f) {
    unsigned u = __float_as_uint(f);
    unsigned r = (u + 0x7FFFu + ((u >> 16) & 1u)) >> 16;
    return (unsigned short)r;
}

// A-matrix fragment (16x32 bf16). lane: M = l&15; element j: K = half*8 + (j>>3)*16 + (j&7)
__device__ __forceinline__ v16bf load_a_frag(const unsigned short* rowptr, int kk, int half) {
    union { v16bf v; uint4 u[2]; } f;
    f.u[0] = *reinterpret_cast<const uint4*>(rowptr + kk + half * 8);
    f.u[1] = *reinterpret_cast<const uint4*>(rowptr + kk + half * 8 + 16);
    return f.v;
}

// B-matrix fragment (32x16 bf16). lane: N = l&15; element j: K = half*8 + (j>>3)*16 + (j&7)
__device__ __forceinline__ v16bf load_b_frag(const unsigned short* rowptr, int kk, int half) {
    union { v16bf v; uint4 u[2]; } f;
    const uint4* p = reinterpret_cast<const uint4*>(rowptr + kk + half * 8);
    f.u[0] = p[0];
    f.u[1] = p[2];
    return f.v;
}

#ifdef HAVE_TDM
// Issue a 2-D bf16 tile DMA: global (row-major, stride_d0 elems) -> LDS.
// D# per CDNA5 ISA. Wave-uniform; tracked by TENSORcnt.
__device__ __forceinline__ void tdm_load_2d(unsigned lds_off, const void* gaddr,
                                            unsigned tile_d0, unsigned tile_d1,
                                            unsigned tensor_d0, unsigned tensor_d1,
                                            unsigned stride_d0) {
    unsigned long long ga = (unsigned long long)(size_t)gaddr;
    u32x4 g0;
    g0[0] = 1u;                                     // count=1, user descriptor
    g0[1] = lds_off;                                // lds_addr (bytes)
    g0[2] = (unsigned)(ga & 0xFFFFFFFFu);           // global_addr[31:0]
    g0[3] = (unsigned)((ga >> 32) & 0x1FFFFFFu) | (2u << 30);  // ga[56:32], type=2
    i32x8 g1;
    g1[0] = (int)(1u << 16);                        // data_size=1 (2 bytes), wg_mask=0
    g1[1] = (int)((tensor_d0 & 0xFFFFu) << 16);     // tensor_dim0[15:0] @ bits63:48
    g1[2] = (int)(((tensor_d0 >> 16) & 0xFFFFu) | ((tensor_d1 & 0xFFFFu) << 16));
    g1[3] = (int)(((tensor_d1 >> 16) & 0xFFFFu) | ((tile_d0 & 0xFFFFu) << 16));
    g1[4] = (int)(tile_d1 & 0xFFFFu);               // tile_dim1; tile_dim2=0
    g1[5] = (int)stride_d0;                         // tensor_dim0_stride[31:0]
    g1[6] = 0;
    g1[7] = 0;
    i32x4 z4 = {0, 0, 0, 0};
#if defined(__clang_major__) && __clang_major__ >= 23
    i32x8 z8 = {0, 0, 0, 0, 0, 0, 0, 0};
    __builtin_amdgcn_tensor_load_to_lds(g0, g1, z4, z4, z8, 0);
#else
    __builtin_amdgcn_tensor_load_to_lds(g0, g1, z4, z4, 0);
#endif
}
#endif

// ---------------------------------------------------------------------------
// Kernel: zero the per-k max buffer
// ---------------------------------------------------------------------------
__global__ void zero6_kernel(unsigned int* p) {
    if (threadIdx.x < KSUB) p[threadIdx.x] = 0u;
}

// ---------------------------------------------------------------------------
// Kernel: fp32 -> bf16 conversion
// ---------------------------------------------------------------------------
__global__ void cvt_kernel(const float* __restrict__ src, unsigned short* __restrict__ dst, int n) {
    int i = blockIdx.x * 256 + threadIdx.x;
    if (i < n) dst[i] = f2bf(src[i]);
}

// ---------------------------------------------------------------------------
// Kernel: per-k global max of mask = max_{n,m} sum_c A[n,c,k]*A[m,c,k]
// A[n,c,k] = w[n,c]*s[n,c,k]; all values >= 0 so float-bit atomicMax is valid.
// ---------------------------------------------------------------------------
__global__ __launch_bounds__(256) void maskmax_kernel(const float* __restrict__ w,
                                                      const float* __restrict__ s,
                                                      unsigned int* __restrict__ maxk) {
    __shared__ float An[16][16][KSUB];
    __shared__ float Am[16][16][KSUB];
    __shared__ float red[256];
    int n0 = blockIdx.x * 16, m0 = blockIdx.y * 16;
    int tid = threadIdx.x;
    for (int i = tid; i < 16 * 16 * KSUB; i += 256) {
        int nloc = i / (16 * KSUB);
        int rem = i % (16 * KSUB);
        int c = rem / KSUB, kk = rem % KSUB;
        An[nloc][c][kk] = w[(n0 + nloc) * 16 + c] * s[((n0 + nloc) * 16 + c) * KSUB + kk];
        Am[nloc][c][kk] = w[(m0 + nloc) * 16 + c] * s[((m0 + nloc) * 16 + c) * KSUB + kk];
    }
    __syncthreads();
    int ty = tid >> 4, tx = tid & 15;
    float d[KSUB] = {0.f, 0.f, 0.f, 0.f, 0.f, 0.f};
    for (int c = 0; c < 16; ++c)
        for (int kk = 0; kk < KSUB; ++kk) d[kk] += An[ty][c][kk] * Am[tx][c][kk];
    for (int kk = 0; kk < KSUB; ++kk) {
        red[tid] = d[kk];
        __syncthreads();
        for (int st = 128; st > 0; st >>= 1) {
            if (tid < st) red[tid] = fmaxf(red[tid], red[tid + st]);
            __syncthreads();
        }
        if (tid == 0) atomicMax((int*)(maxk + kk), __float_as_int(red[0]));
        __syncthreads();
    }
}

// ---------------------------------------------------------------------------
// Kernel: projection GEMM  out = (Xbf @ Wbf^T + bias) * scale  (bf16 out)
// ---------------------------------------------------------------------------
__global__ __launch_bounds__(128) void proj_kernel(const unsigned short* __restrict__ Xbf,
                                                   const unsigned short* __restrict__ Wbf,
                                                   const float* __restrict__ bias,
                                                   unsigned short* __restrict__ outb,
                                                   float scale) {
    int tid = threadIdx.x;
    int wave = tid >> 5, lane = tid & 31;
    int half = lane >> 4, ln16 = lane & 15;
    int n0 = blockIdx.x * 16;
    int col0 = blockIdx.y * 64 + wave * 16;

    v8f acc = {0.f, 0.f, 0.f, 0.f, 0.f, 0.f, 0.f, 0.f};
    const unsigned short* arow = Xbf + (size_t)(n0 + ln16) * HDIM;
    const unsigned short* brow = Wbf + (size_t)(col0 + ln16) * HDIM;
    for (int kk = 0; kk < HDIM; kk += 32) {
        v16bf a = load_a_frag(arow, kk, half);
        v16bf b = load_b_frag(brow, kk, half);
        acc = __builtin_amdgcn_wmma_f32_16x16x32_bf16(false, a, false, b, (short)0, acc, false, false);
    }
    for (int r = 0; r < 8; ++r) {
        int m = half * 8 + r;
        int h = col0 + ln16;
        float v = (acc[r] + bias[h]) * scale;
        outb[(size_t)(n0 + m) * HDIM + h] = f2bf(v);
    }
}

// ---------------------------------------------------------------------------
// Fused flash-attention kernel with on-the-fly log mask + Wo GEMM + LN.
// Grid: (N/16, K). Block: 128 threads (4 waves). Q/K tiles staged by the
// Tensor Data Mover (wave 0 issues, TENSORcnt fence), V transposed in LDS.
// Dynamic LDS carving (87 KB), phase-aliased:
//   [0      , 16384) Qb  (16x512 bf16)        | phase2: ctx bf16
//   [16384  , 49152) Kb  (32x512 bf16)        | phase2: ctx f32
//   [49152  , 81920) Vt  (512x32 bf16, V^T)   | phase2: outpre f32
//   [81920  , 82944) Aq  (16x16 f32)          | phase2: LN scratch
//   [82944  , 84992) Akt (32x16 f32)
//   [84992  , 89088) Pb  (4 waves x 16x32 bf16)
// ---------------------------------------------------------------------------
__global__ __launch_bounds__(128) void attn_kernel(const float* __restrict__ X,
                                                   const unsigned short* __restrict__ qbuf,
                                                   const unsigned short* __restrict__ kbuf,
                                                   const unsigned short* __restrict__ vbuf,
                                                   const unsigned short* __restrict__ Wobf,
                                                   const float* __restrict__ w,
                                                   const float* __restrict__ s,
                                                   const unsigned int* __restrict__ maxku,
                                                   const float* __restrict__ bo,
                                                   const float* __restrict__ gamma,
                                                   const float* __restrict__ beta,
                                                   float* __restrict__ out) {
    extern __shared__ char smem[];
    unsigned short* Qb  = (unsigned short*)(smem);
    unsigned short* Kb  = (unsigned short*)(smem + 16384);
    unsigned short* Vt  = (unsigned short*)(smem + 49152);
    float*          Aq  = (float*)(smem + 81920);
    float*          Akt = (float*)(smem + 82944);
    unsigned short* Pb  = (unsigned short*)(smem + 84992);

    int tid = threadIdx.x;
    int wave = tid >> 5, lane = tid & 31;
    int half = lane >> 4, ln16 = lane & 15;
    int nblk = blockIdx.x, kidx = blockIdx.y;
    int qrow0 = nblk * 16;

#ifdef HAVE_TDM
    unsigned Qb_off = (unsigned)(size_t)Qb;   // generic LDS addr low 32 bits = LDS offset
    unsigned Kb_off = (unsigned)(size_t)Kb;
#endif

    // ---- stage Q rows (bf16) and Aq ----
#ifdef HAVE_TDM
    if (wave == 0) {
        tdm_load_2d(Qb_off, qbuf + (size_t)qrow0 * HDIM, HDIM, 16, HDIM, NSTK, HDIM);
    }
#else
    {
        const uint4* qsrc = (const uint4*)(qbuf + (size_t)qrow0 * HDIM);
        uint4* qdst = (uint4*)Qb;
        for (int i = tid; i < 1024; i += 128) qdst[i] = qsrc[i];
    }
#endif
    for (int i = tid; i < 256; i += 128) {
        int m = i >> 4, c = i & 15;
        Aq[i] = w[(qrow0 + m) * 16 + c] * s[((qrow0 + m) * 16 + c) * KSUB + kidx];
    }
    float inv_mk = 1.0f / (__uint_as_float(maxku[kidx]) + 1e-8f);

    v8f Oa[8];
    for (int tt = 0; tt < 8; ++tt) Oa[tt] = (v8f){0.f, 0.f, 0.f, 0.f, 0.f, 0.f, 0.f, 0.f};
    float m_i[8], l_i[8];
    for (int r = 0; r < 8; ++r) { m_i[r] = -1e30f; l_i[r] = 0.f; }

#ifdef HAVE_TDM
    if (wave == 0) __builtin_amdgcn_s_wait_tensorcnt(0);
#endif
    __syncthreads();

    for (int j0 = 0; j0 < NSTK; j0 += 32) {
        __syncthreads();  // protect Kb/Vt/Akt overwrite
        // K tile row-major via TDM (async DMA, TENSORcnt-tracked)
#ifdef HAVE_TDM
        if (wave == 0) {
            tdm_load_2d(Kb_off, kbuf + (size_t)j0 * HDIM, HDIM, 32, HDIM, NSTK, HDIM);
        }
#else
        {
            const uint4* ks = (const uint4*)(kbuf + (size_t)j0 * HDIM);
            uint4* kd = (uint4*)Kb;
            for (int i = tid; i < 2048; i += 128) kd[i] = ks[i];
        }
#endif
        // V tile transposed: Vt[col][key]; pack key pairs into dword stores
        for (int i = tid; i < 1024; i += 128) {
            int kp = i >> 6;   // key pair 0..15 -> keys 2kp, 2kp+1
            int q4 = i & 63;   // 8-col group
            uint4 d0 = ((const uint4*)(vbuf + (size_t)(j0 + 2 * kp) * HDIM))[q4];
            uint4 d1 = ((const uint4*)(vbuf + (size_t)(j0 + 2 * kp + 1) * HDIM))[q4];
            const unsigned short* p0 = (const unsigned short*)&d0;
            const unsigned short* p1 = (const unsigned short*)&d1;
            int c8 = q4 * 8;
            unsigned* vt32 = (unsigned*)Vt;
            for (int e = 0; e < 8; ++e)
                vt32[(((c8 + e) * 32) + 2 * kp) >> 1] = (unsigned)p0[e] | ((unsigned)p1[e] << 16);
        }
        // Akt
        for (int i = tid; i < 512; i += 128) {
            int mloc = i >> 4, c = i & 15;
            int mg = j0 + mloc;
            Akt[i] = w[mg * 16 + c] * s[(mg * 16 + c) * KSUB + kidx];
        }
#ifdef HAVE_TDM
        if (wave == 0) __builtin_amdgcn_s_wait_tensorcnt(0);
#endif
        __syncthreads();

        // --- scores + mask: two 16x16 tiles covering 32 keys ---
        v8f S[2];
        for (int t = 0; t < 2; ++t) {
            v8f acc = {0.f, 0.f, 0.f, 0.f, 0.f, 0.f, 0.f, 0.f};
            const unsigned short* qrow = Qb + ln16 * HDIM;
            const unsigned short* krow = Kb + (t * 16 + ln16) * HDIM;
            for (int kk = 0; kk < HDIM; kk += 32) {
                v16bf a = load_a_frag(qrow, kk, half);
                v16bf b = load_b_frag(krow, kk, half);
                acc = __builtin_amdgcn_wmma_f32_16x16x32_bf16(false, a, false, b, (short)0, acc, false, false);
            }
            // rank-16 mask dot via fp32 WMMA (16x16x4), 4 steps
            v8f macc = {0.f, 0.f, 0.f, 0.f, 0.f, 0.f, 0.f, 0.f};
#if __has_builtin(__builtin_amdgcn_wmma_f32_16x16x4_f32)
            for (int c0 = 0; c0 < 16; c0 += 4) {
                v2f a, b;
                a.x = Aq[ln16 * 16 + c0 + half * 2];
                a.y = Aq[ln16 * 16 + c0 + half * 2 + 1];
                b.x = Akt[(t * 16 + ln16) * 16 + c0 + half * 2];
                b.y = Akt[(t * 16 + ln16) * 16 + c0 + half * 2 + 1];
                macc = __builtin_amdgcn_wmma_f32_16x16x4_f32(false, a, false, b, (short)0, macc, false, false);
            }
#else
            for (int r = 0; r < 8; ++r) {
                int mrow = half * 8 + r;
                float acc2 = 0.f;
                for (int c = 0; c < 16; ++c)
                    acc2 += Aq[mrow * 16 + c] * Akt[(t * 16 + ln16) * 16 + c];
                macc[r] = acc2;
            }
#endif
            for (int r = 0; r < 8; ++r)
                S[t][r] = acc[r] + __logf(macc[r] * inv_mk + 1e-10f);
        }

        // --- online softmax (rows in VGPR slot r, spread over 16 lanes/half) ---
        float newm[8], alpha[8];
        for (int r = 0; r < 8; ++r) {
            float v = fmaxf(S[0][r], S[1][r]);
            for (int msk = 8; msk >= 1; msk >>= 1) v = fmaxf(v, __shfl_xor(v, msk, 32));
            newm[r] = fmaxf(m_i[r], v);
            alpha[r] = __expf(m_i[r] - newm[r]);
            m_i[r] = newm[r];
        }
        for (int r = 0; r < 8; ++r) {
            float p0 = __expf(S[0][r] - newm[r]);
            float p1 = __expf(S[1][r] - newm[r]);
            S[0][r] = p0;
            S[1][r] = p1;
            float rs = p0 + p1;
            for (int msk = 8; msk >= 1; msk >>= 1) rs += __shfl_xor(rs, msk, 32);
            l_i[r] = l_i[r] * alpha[r] + rs;
        }
        for (int tt = 0; tt < 8; ++tt)
            for (int r = 0; r < 8; ++r) Oa[tt][r] *= alpha[r];

        // --- stage P (D layout -> row-major bf16) into per-wave LDS region ---
        unsigned short* Pw = Pb + wave * 512;
        for (int t = 0; t < 2; ++t)
            for (int r = 0; r < 8; ++r)
                Pw[(half * 8 + r) * 32 + t * 16 + ln16] = f2bf(S[t][r]);
        __asm volatile("" ::: "memory");  // wave-local LDS RAW: keep program order

        // --- PV: one 16x16x32 WMMA per 16-col output tile (8 tiles = 128 cols) ---
        v16bf pa = load_a_frag(Pw + ln16 * 32, 0, half);
        for (int tt = 0; tt < 8; ++tt) {
            int col0 = wave * 128 + tt * 16;
            v16bf vb = load_b_frag(Vt + (col0 + ln16) * 32, 0, half);
            Oa[tt] = __builtin_amdgcn_wmma_f32_16x16x32_bf16(false, pa, false, vb, (short)0, Oa[tt], false, false);
        }
    }

    // ---- epilogue: ctx = O / l, then out = ctx @ Wo^T + bo + X, LayerNorm ----
    float* ctxf = (float*)(smem + 16384);  // alias Kb
    for (int r = 0; r < 8; ++r) {
        float inv_l = 1.0f / l_i[r];
        for (int tt = 0; tt < 8; ++tt) {
            int col0 = wave * 128 + tt * 16;
            ctxf[(half * 8 + r) * HDIM + col0 + ln16] = Oa[tt][r] * inv_l;
        }
    }
    __syncthreads();
    unsigned short* ctxbf = (unsigned short*)smem;  // alias Qb
    for (int i = tid; i < 16 * HDIM; i += 128) ctxbf[i] = f2bf(ctxf[i]);
    __syncthreads();

    float* outpre = (float*)(smem + 49152);  // alias Vt
    for (int tt = 0; tt < 8; ++tt) {
        int col0 = wave * 128 + tt * 16;
        v8f acc = {0.f, 0.f, 0.f, 0.f, 0.f, 0.f, 0.f, 0.f};
        const unsigned short* crow = ctxbf + ln16 * HDIM;
        const unsigned short* brow = Wobf + (size_t)(col0 + ln16) * HDIM;
        for (int kk = 0; kk < HDIM; kk += 32) {
            v16bf a = load_a_frag(crow, kk, half);
            v16bf b = load_b_frag(brow, kk, half);
            acc = __builtin_amdgcn_wmma_f32_16x16x32_bf16(false, a, false, b, (short)0, acc, false, false);
        }
        for (int r = 0; r < 8; ++r) {
            int m = half * 8 + r;
            int h = col0 + ln16;
            outpre[m * HDIM + h] = acc[r] + bo[h] + X[(size_t)(qrow0 + m) * HDIM + h];
        }
    }
    __syncthreads();

    // LayerNorm over H per row (eps=1e-12), affine by (gamma,beta)[kidx]
    float* lnS = (float*)(smem + 81920);
    float* lnS2 = lnS + 128;
    float* lmu = lnS + 256;
    float* lrs = lnS + 272;
    {
        int row = tid >> 3, p = tid & 7;
        int c0 = p * 64;
        float sm = 0.f, s2 = 0.f;
        for (int e = 0; e < 64; ++e) {
            float x = outpre[row * HDIM + c0 + e];
            sm += x;
            s2 += x * x;
        }
        lnS[tid] = sm;
        lnS2[tid] = s2;
    }
    __syncthreads();
    if ((tid & 7) == 0) {
        int row = tid >> 3;
        float sm = 0.f, s2 = 0.f;
        for (int e = 0; e < 8; ++e) {
            sm += lnS[row * 8 + e];
            s2 += lnS2[row * 8 + e];
        }
        float mu = sm * (1.0f / HDIM);
        float var = s2 * (1.0f / HDIM) - mu * mu;
        lmu[row] = mu;
        lrs[row] = rsqrtf(var + 1e-12f);
    }
    __syncthreads();
    {
        int row = tid >> 3, p = tid & 7;
        int c0 = p * 64;
        float mu = lmu[row], rs = lrs[row];
        float* orow = out + ((size_t)kidx * NSTK + qrow0 + row) * HDIM;
        const float* g = gamma + kidx * HDIM;
        const float* bt = beta + kidx * HDIM;
        for (int e = 0; e < 64; ++e) {
            int h = c0 + e;
            float x = outpre[row * HDIM + h];
            orow[h] = (x - mu) * rs * g[h] + bt[h];
        }
    }
}

// ---------------------------------------------------------------------------
// Host launcher
// ---------------------------------------------------------------------------
extern "C" void kernel_launch(void* const* d_in, const int* in_sizes, int n_in,
                              void* d_out, int out_size, void* d_ws, size_t ws_size,
                              hipStream_t stream) {
    const float* X  = (const float*)d_in[0];
    const float* w  = (const float*)d_in[1];
    const float* s  = (const float*)d_in[2];
    const float* Wq = (const float*)d_in[4];
    const float* bq = (const float*)d_in[5];
    const float* Wk = (const float*)d_in[6];
    const float* bk = (const float*)d_in[7];
    const float* Wv = (const float*)d_in[8];
    const float* bv = (const float*)d_in[9];
    const float* Wo = (const float*)d_in[10];
    const float* bo = (const float*)d_in[11];
    const float* gamma = (const float*)d_in[12];
    const float* beta  = (const float*)d_in[13];
    float* out = (float*)d_out;

    char* ws = (char*)d_ws;
    unsigned short* Xbf  = (unsigned short*)(ws + 0);         // 4 MB
    unsigned short* Wqbf = (unsigned short*)(ws + 4194304);   // 512 KB
    unsigned short* Wkbf = (unsigned short*)(ws + 4718592);
    unsigned short* Wvbf = (unsigned short*)(ws + 5242880);
    unsigned short* Wobf = (unsigned short*)(ws + 5767168);
    unsigned short* qbuf = (unsigned short*)(ws + 6291456);   // 4 MB
    unsigned short* kbuf = (unsigned short*)(ws + 10485760);  // 4 MB
    unsigned short* vbuf = (unsigned short*)(ws + 14680064);  // 4 MB
    unsigned int*   maxk = (unsigned int*)(ws + 18874368);

    zero6_kernel<<<1, 32, 0, stream>>>(maxk);
    cvt_kernel<<<(NSTK * HDIM + 255) / 256, 256, 0, stream>>>(X, Xbf, NSTK * HDIM);
    cvt_kernel<<<(HDIM * HDIM + 255) / 256, 256, 0, stream>>>(Wq, Wqbf, HDIM * HDIM);
    cvt_kernel<<<(HDIM * HDIM + 255) / 256, 256, 0, stream>>>(Wk, Wkbf, HDIM * HDIM);
    cvt_kernel<<<(HDIM * HDIM + 255) / 256, 256, 0, stream>>>(Wv, Wvbf, HDIM * HDIM);
    cvt_kernel<<<(HDIM * HDIM + 255) / 256, 256, 0, stream>>>(Wo, Wobf, HDIM * HDIM);

    maskmax_kernel<<<dim3(NSTK / 16, NSTK / 16), 256, 0, stream>>>(w, s, maxk);

    const float qscale = 0.044194173824159216f;  // 1/sqrt(512)
    proj_kernel<<<dim3(NSTK / 16, HDIM / 64), 128, 0, stream>>>(Xbf, Wqbf, bq, qbuf, qscale);
    proj_kernel<<<dim3(NSTK / 16, HDIM / 64), 128, 0, stream>>>(Xbf, Wkbf, bk, kbuf, 1.0f);
    proj_kernel<<<dim3(NSTK / 16, HDIM / 64), 128, 0, stream>>>(Xbf, Wvbf, bv, vbuf, 1.0f);

    attn_kernel<<<dim3(NSTK / 16, KSUB), 128, 89088, stream>>>(
        X, qbuf, kbuf, vbuf, Wobf, w, s, maxk, bo, gamma, beta, out);
}
